// NodeModelBP_single_50242527429374
// MI455X (gfx1250) — matrix operand order repaired and checked by reading, not compiled, not run on hardware
//
#include <hip/hip_runtime.h>
#include <hip/hip_bf16.h>

// ---------------------------------------------------------------------------
// NodeModelBP fused pipeline for MI455X (gfx1250, wave32, WMMA)
//   K1: zero agg workspace                (d_ws)
//   K2: scatter-add edge_attr -> agg      (global_atomic_add_f32, BW-bound)
//   K3: gather+concat + MLP via v_wmma_f32_16x16x32_f16
// ---------------------------------------------------------------------------

#define N_NODES   100000
#define N_EDGES   1600000
#define N_AGENTS  50000
#define D_X       64
#define D_LSTM    64
#define D_Z       32
#define D_EDGE    48
#define D_IN      208          // 64+64+32+48
#define KP        224          // D_IN padded to multiple of 32 (7 k-steps)
#define D_HID     256
#define D_OUT     64

typedef __attribute__((ext_vector_type(16))) _Float16 v16h;
typedef __attribute__((ext_vector_type(8)))  _Float16 v8h;
typedef __attribute__((ext_vector_type(8)))  float    v8f;

// ---------------------------------------------------------------------------
// K1: zero the aggregation buffer (100000 x 48 f32) with float4 stores
// ---------------------------------------------------------------------------
__global__ __launch_bounds__(256) void k_zero_agg(float4* agg4, int n4) {
    int i = blockIdx.x * blockDim.x + threadIdx.x;
    if (i < n4) agg4[i] = make_float4(0.f, 0.f, 0.f, 0.f);
}

// ---------------------------------------------------------------------------
// K2: scatter-sum. One thread per (edge, channel) element; coalesced reads of
// edge_attr, atomic f32 adds into agg[row[e]*48 + c].
// ---------------------------------------------------------------------------
__global__ __launch_bounds__(256) void k_scatter_add(const float* __restrict__ edge_attr,
                                                     const int*   __restrict__ edge_row,
                                                     float*       __restrict__ agg) {
    long long i = (long long)blockIdx.x * blockDim.x + threadIdx.x;
    const long long total = (long long)N_EDGES * D_EDGE;
    if (i >= total) return;
    int e = (int)(i / D_EDGE);
    int c = (int)(i % D_EDGE);
    int node = edge_row[e];                    // edge_index[0][e]
    atomicAdd(&agg[(long long)node * D_EDGE + c], edge_attr[i]);
}

// ---------------------------------------------------------------------------
// WMMA fragment loaders from LDS (row-major in K, 16B-aligned).
// A-matrix 16x32 f16 layout (ISA 7.12.2): per lane, elems 0-7 = K kb..kb+7,
// elems 8-15 = K kb+16..kb+23 (kb already includes the lane>=16 +8 shift).
// ---------------------------------------------------------------------------
__device__ __forceinline__ v16h lds_frag_a(const _Float16* rowBase, int kb) {
    v8h lo = *(const v8h*)(rowBase + kb);
    v8h hi = *(const v8h*)(rowBase + kb + 16);
    v16h r;
#pragma unroll
    for (int j = 0; j < 8; ++j) { r[j] = lo[j]; r[j + 8] = hi[j]; }
    return r;
}

// B-matrix 32x16 f16: lanes hold one column n; elems 0-7 = K kb..kb+7,
// elems 8-15 = K kb+16..kb+23 (kb includes the lane>=16 +8 shift). Weights stored transposed [N][K].
__device__ __forceinline__ v16h lds_frag_b(const _Float16* rowBase, int kb) {
    v8h lo = *(const v8h*)(rowBase + kb);
    v8h hi = *(const v8h*)(rowBase + kb + 16);
    v16h r;
#pragma unroll
    for (int j = 0; j < 8; ++j) { r[j] = lo[j]; r[j + 8] = hi[j]; }
    return r;
}

// ---------------------------------------------------------------------------
// K3: fused gather + concat + MLP.
// Block: 256 threads (8 wave32), M-tile = 64 agents.
// GEMM1: [64 x 224] x [224 x 256] -> ReLU -> H [64 x 256]
// GEMM2: [64 x 256] x [256 x 64]  -> out
// Dynamic LDS: W1^T [256][224] | W2^T [64][256] | A [64][224] | H [64][256]
// ---------------------------------------------------------------------------
__global__ __launch_bounds__(256) void k_mlp_wmma(
        const float* __restrict__ x,
        const float* __restrict__ x_lstm,
        const float* __restrict__ z,
        const float* __restrict__ agg,
        const int*   __restrict__ node_idx,
        const float* __restrict__ W1, const float* __restrict__ b1,
        const float* __restrict__ W2, const float* __restrict__ b2,
        float* __restrict__ out) {
    extern __shared__ _Float16 smem[];
    _Float16* sW1t = smem;                       // 256*KP halfs
    _Float16* sW2t = sW1t + D_HID * KP;          // 64*256 halfs
    _Float16* sA   = sW2t + D_OUT * D_HID;       // 64*KP  halfs
    _Float16* sH   = sA   + 64 * KP;             // 64*256 halfs

    const int tid     = threadIdx.x;
    const int rowBase = blockIdx.x * 64;

    // ---- stage W1 transposed, f32 -> f16, K padded with zeros ----
    for (int i = tid; i < D_HID * KP; i += 256) {
        int n = i / KP, k = i % KP;
        sW1t[i] = (k < D_IN) ? (_Float16)W1[k * D_HID + n] : (_Float16)0.f;
    }
    // ---- stage W2 transposed ----
    for (int i = tid; i < D_OUT * D_HID; i += 256) {
        int n = i / D_HID, k = i % D_HID;
        sW2t[i] = (_Float16)W2[k * D_OUT + n];
    }
    // ---- gather + concat into A tile [64][KP] ----
    for (int i = tid; i < 64 * KP; i += 256) {
        int r = i / KP, c = i % KP;
        int a = rowBase + r;
        float v = 0.f;
        if (a < N_AGENTS && c < D_IN) {
            if (c < D_X) {
                int nd = node_idx[a];
                v = x[(long long)nd * D_X + c];
            } else if (c < D_X + D_LSTM) {
                v = x_lstm[(long long)a * D_LSTM + (c - D_X)];
            } else if (c < D_X + D_LSTM + D_Z) {
                v = z[(long long)a * D_Z + (c - D_X - D_LSTM)];
            } else {
                int nd = node_idx[a];
                v = agg[(long long)nd * D_EDGE + (c - D_X - D_LSTM - D_Z)];
            }
        }
        sA[i] = (_Float16)v;
    }
    __syncthreads();

    const int wave    = tid >> 5;
    const int lane    = tid & 31;
    const int lane16  = lane & 15;
    const int hiHalf  = lane >> 4;       // 0 or 1
    const int aKShift = hiHalf * 8;      // A-frag K shift for lanes 16-31
    const int bKShift = hiHalf * 8;      // B-frag K shift for lanes 16-31

    // ================= GEMM1: A[64xKP] * W1[KPx256] =================
    const int mt    = wave >> 1;                 // 0..3
    const int nBase = (wave & 1) * 128;          // 0 or 128
    v8f acc[8] = {};
    const _Float16* aRow = sA + (mt * 16 + lane16) * KP;
#pragma unroll
    for (int kt = 0; kt < KP / 32; ++kt) {
        v16h af = lds_frag_a(aRow, kt * 32 + aKShift);
#pragma unroll
        for (int t = 0; t < 8; ++t) {
            const _Float16* bRow = sW1t + (nBase + t * 16 + lane16) * KP;
            v16h bf = lds_frag_b(bRow, kt * 32 + bKShift);
            acc[t] = __builtin_amdgcn_wmma_f32_16x16x32_f16(
                false, af, false, bf, (short)0, acc[t], false, false);
        }
    }
    // bias + ReLU -> H (f16, row-major [64][256])
#pragma unroll
    for (int t = 0; t < 8; ++t) {
        int n = nBase + t * 16 + lane16;
        float bias = b1[n];
        int mBase = mt * 16 + hiHalf * 8;
#pragma unroll
        for (int r = 0; r < 8; ++r) {
            float h = acc[t][r] + bias;
            h = h > 0.f ? h : 0.f;
            sH[(mBase + r) * D_HID + n] = (_Float16)h;
        }
    }
    __syncthreads();

    // ================= GEMM2: H[64x256] * W2[256x64] =================
    const int mt2     = wave >> 1;               // 0..3
    const int ntBase  = (wave & 1) * 2;          // tile pair: n = 0/16 or 32/48
    v8f acc2[2] = {};
    const _Float16* aRow2 = sH + (mt2 * 16 + lane16) * D_HID;
#pragma unroll
    for (int kt = 0; kt < D_HID / 32; ++kt) {
        v16h af = lds_frag_a(aRow2, kt * 32 + aKShift);
#pragma unroll
        for (int t = 0; t < 2; ++t) {
            const _Float16* bRow = sW2t + ((ntBase + t) * 16 + lane16) * D_HID;
            v16h bf = lds_frag_b(bRow, kt * 32 + bKShift);
            acc2[t] = __builtin_amdgcn_wmma_f32_16x16x32_f16(
                false, af, false, bf, (short)0, acc2[t], false, false);
        }
    }
    // bias + store f32 output [N_AGENTS][64]
#pragma unroll
    for (int t = 0; t < 2; ++t) {
        int n = (ntBase + t) * 16 + lane16;
        float bias = b2[n];
        int mBase = mt2 * 16 + hiHalf * 8;
#pragma unroll
        for (int r = 0; r < 8; ++r) {
            int a = rowBase + mBase + r;
            if (a < N_AGENTS) out[(long long)a * D_OUT + n] = acc2[t][r] + bias;
        }
    }
}

// ---------------------------------------------------------------------------
extern "C" void kernel_launch(void* const* d_in, const int* in_sizes, int n_in,
                              void* d_out, int out_size, void* d_ws, size_t ws_size,
                              hipStream_t stream) {
    const float* x        = (const float*)d_in[0];
    const float* x_lstm   = (const float*)d_in[1];
    const float* z        = (const float*)d_in[2];
    const int*   edge_idx = (const int*)  d_in[3];  // [2][N_EDGES]; row = first half
    const float* edge_attr= (const float*)d_in[4];
    const int*   node_idx = (const int*)  d_in[5];
    const float* W1       = (const float*)d_in[6];
    const float* b1       = (const float*)d_in[7];
    const float* W2       = (const float*)d_in[8];
    const float* b2       = (const float*)d_in[9];
    float*       out      = (float*)d_out;

    // agg scratch: 100000 x 48 f32 = 19.2 MB in d_ws
    float* agg = (float*)d_ws;

    // K1: zero agg
    {
        int n4 = (N_NODES * D_EDGE) / 4;
        int blocks = (n4 + 255) / 256;
        k_zero_agg<<<blocks, 256, 0, stream>>>((float4*)agg, n4);
    }
    // K2: scatter-add
    {
        long long total = (long long)N_EDGES * D_EDGE;
        int blocks = (int)((total + 255) / 256);
        k_scatter_add<<<blocks, 256, 0, stream>>>(edge_attr, edge_idx, agg);
    }
    // K3: fused gather + MLP with WMMA
    {
        int blocks = (N_AGENTS + 63) / 64;  // 782
        size_t shmem = (size_t)(D_HID * KP + D_OUT * D_HID + 64 * KP + 64 * D_HID)
                       * sizeof(_Float16);  // ~204 KB (<= 320 KB WGP LDS)
        k_mlp_wmma<<<blocks, 256, shmem, stream>>>(
            x, x_lstm, z, agg, node_idx, W1, b1, W2, b2, out);
    }
}
